// OuterProductMean_61400852464431
// MI455X (gfx1250) — matrix operand order repaired and checked by reading, not compiled, not run on hardware
//
#include <hip/hip_runtime.h>
#include <hip/hip_bf16.h>

typedef __attribute__((ext_vector_type(16))) __bf16 v16bf;
typedef __attribute__((ext_vector_type(8)))  __bf16 v8bf;
typedef __attribute__((ext_vector_type(8)))  float  v8f;

#define NM 512     // MSA rows (m)
#define NN 256     // residues (i/j)
#define NC 256     // input channels
#define NMID 32    // mid channels
#define NOUT 128   // output channels
#define NP 8192    // NN*NMID

// ---------------------------------------------------------------------------
// Kernel 1: wo [1024,128] f32 -> wo_t [128][1024] bf16 (K-contiguous rows)
// ---------------------------------------------------------------------------
__global__ void prep_wo_kernel(const float* __restrict__ wo, __bf16* __restrict__ wo_t) {
    int idx = blockIdx.x * blockDim.x + threadIdx.x;   // 0 .. 128*1024-1
    int o  = idx >> 10;
    int kk = idx & 1023;
    wo_t[(size_t)o * 1024 + kk] = (__bf16)wo[(size_t)kk * NOUT + o];
}

// ---------------------------------------------------------------------------
// Kernel 2: norm[i,j] = sum_m mask[m,i]*mask[m,j]
// ---------------------------------------------------------------------------
__global__ void norm_kernel(const float* __restrict__ mask, float* __restrict__ nrm) {
    int i = blockIdx.x;
    int j = threadIdx.x;
    float acc = 0.f;
    #pragma unroll 4
    for (int m = 0; m < NM; ++m)
        acc += mask[m * NN + i] * mask[m * NN + j];
    nrm[i * NN + j] = acc;
}

// ---------------------------------------------------------------------------
// Kernel 3: LayerNorm + dual projection + mask.
// One wave (32 lanes) per (m,n) row. Writes Lbuf/Rbuf transposed: [p][m] bf16,
// p = n*32 + x, so the GEMM kernel gets contiguous K (=m) per matrix row.
// ---------------------------------------------------------------------------
__global__ __launch_bounds__(256) void ln_proj_kernel(
    const float* __restrict__ act, const float* __restrict__ mask,
    const float* __restrict__ ln_g, const float* __restrict__ ln_b,
    const float* __restrict__ wl, const float* __restrict__ bl,
    const float* __restrict__ wr, const float* __restrict__ br,
    __bf16* __restrict__ Lbuf, __bf16* __restrict__ Rbuf)
{
    __shared__ float rowbuf[8][NC];   // one normalized row per wave
    const int lane = threadIdx.x & 31;
    const int wave = threadIdx.x >> 5;

    #pragma unroll 1
    for (int it = 0; it < 4; ++it) {
        const int r = blockIdx.x * 32 + it * 8 + wave;   // row id in [0, M*N)
        const int m = r >> 8;
        const int n = r & 255;
        const size_t base = (size_t)r * NC;

        // ---- mean / var over 256 channels, 8 per lane (c = k*32 + lane) ----
        float vals[8];
        float s = 0.f, s2 = 0.f;
        #pragma unroll
        for (int k = 0; k < 8; ++k) {
            float v = act[base + k * 32 + lane];
            vals[k] = v; s += v; s2 += v * v;
        }
        #pragma unroll
        for (int off = 16; off >= 1; off >>= 1) {
            s  += __shfl_xor(s,  off, 32);
            s2 += __shfl_xor(s2, off, 32);
        }
        const float mean = s * (1.f / NC);
        const float var  = s2 * (1.f / NC) - mean * mean;
        const float rstd = rsqrtf(var + 1e-5f);

        #pragma unroll
        for (int k = 0; k < 8; ++k) {
            int c = k * 32 + lane;
            rowbuf[wave][c] = (vals[k] - mean) * rstd * ln_g[c] + ln_b[c];
        }
        // same-wave LDS ops are in order; compiler inserts s_wait_dscnt

        // ---- dual projection: lane 'x' computes left[x], right[x] ----
        float aL = bl[lane], aR = br[lane];
        #pragma unroll 4
        for (int c = 0; c < NC; ++c) {
            float v = rowbuf[wave][c];
            aL += v * wl[c * NMID + lane];
            aR += v * wr[c * NMID + lane];
        }
        const float mv = mask[r];
        const size_t pi = (size_t)(n * NMID + lane) * NM + m;
        Lbuf[pi] = (__bf16)(mv * aL);
        Rbuf[pi] = (__bf16)(mv * aR);
    }
}

// ---------------------------------------------------------------------------
// Kernel 4: fused outer-product GEMM (K=512) + wo projection (K=1024) + norm.
// Block = 4x4 (i,j) pairs => 128x128 C tile. 8 waves: 2(prow) x 4(qcol),
// each wave computes a 64x32 strip (4x2 WMMA fragments).
// ---------------------------------------------------------------------------
__global__ __launch_bounds__(256, 2) void opm_gemm_kernel(
    const __bf16* __restrict__ Lbuf, const __bf16* __restrict__ Rbuf,
    const __bf16* __restrict__ wo_t, const float* __restrict__ bo,
    const float* __restrict__ nrm, float* __restrict__ out)
{
    __shared__ __bf16 Xls[16 * 1024];   // 16 (i,j) pairs x 1024 (x,y), bf16

    const int lane = threadIdx.x & 31;
    const int wave = threadIdx.x >> 5;
    const int wrow = wave >> 2;          // 0..1
    const int wcol = wave & 3;           // 0..3
    const int lm   = lane & 15;
    const int hi   = (lane >> 4) & 1;
    const int ka   = hi ? 16 : 0;        // A-frag chunk base (contiguous K)
    const int kb   = hi ? 16 : 0;        // B-frag chunk base

    const int ibase = blockIdx.y * 4;
    const int jbase = blockIdx.x * 4;
    const int pbase = ibase * NMID;
    const int qbase = jbase * NMID;

    v8f acc[4][2];
    #pragma unroll
    for (int rt = 0; rt < 4; ++rt)
        #pragma unroll
        for (int ct = 0; ct < 2; ++ct)
            acc[rt][ct] = (v8f){0.f,0.f,0.f,0.f,0.f,0.f,0.f,0.f};

    const __bf16* aptr[4];
    #pragma unroll
    for (int rt = 0; rt < 4; ++rt)
        aptr[rt] = Lbuf + (size_t)(pbase + wrow * 64 + rt * 16 + lm) * NM;
    const __bf16* bptr[2];
    #pragma unroll
    for (int ct = 0; ct < 2; ++ct)
        bptr[ct] = Rbuf + (size_t)(qbase + wcol * 32 + ct * 16 + lm) * NM;

    // ---- stage A: C[128x128] = L^T L-strip . R-strip, K = 512 ----
    #pragma unroll 2
    for (int kt = 0; kt < 16; ++kt) {
        const int k0 = kt * 32;
        __builtin_prefetch(bptr[0] + k0 + 64, 0, 0);   // -> global_prefetch_b8
        __builtin_prefetch(aptr[0] + k0 + 64, 0, 0);
        v16bf bfr[2];
        #pragma unroll
        for (int ct = 0; ct < 2; ++ct)
            bfr[ct] = *(const v16bf*)(bptr[ct] + k0 + kb);   // 16 contig K
        #pragma unroll
        for (int rt = 0; rt < 4; ++rt) {
            v8bf lo = *(const v8bf*)(aptr[rt] + k0 + ka);
            v8bf hg = *(const v8bf*)(aptr[rt] + k0 + ka + 8);
            v16bf a = __builtin_shufflevector(lo, hg,
                0,1,2,3,4,5,6,7,8,9,10,11,12,13,14,15);
            #pragma unroll
            for (int ct = 0; ct < 2; ++ct)
                acc[rt][ct] = __builtin_amdgcn_wmma_f32_16x16x32_bf16(
                    false, a, false, bfr[ct], (short)0, acc[rt][ct], false, false);
        }
    }

    // ---- C tile -> LDS as bf16, indexed [pair=(i_loc*4+j_loc)][x*32+y] ----
    #pragma unroll
    for (int rt = 0; rt < 4; ++rt) {
        #pragma unroll
        for (int ct = 0; ct < 2; ++ct) {
            const int q_loc = wcol * 32 + ct * 16 + lm;
            const int j_loc = q_loc >> 5, y = q_loc & 31;
            #pragma unroll
            for (int v = 0; v < 8; ++v) {
                const int p_loc = wrow * 64 + rt * 16 + hi * 8 + v;
                const int i_loc = p_loc >> 5, x = p_loc & 31;
                Xls[(((i_loc << 2) | j_loc) << 10) | (x << 5) | y] =
                    (__bf16)acc[rt][ct][v];
            }
        }
    }
    __syncthreads();

    // ---- stage B: [16 pairs x 1024] @ wo_t^T -> [16 x 128], K = 1024 ----
    v8f po = (v8f){0.f,0.f,0.f,0.f,0.f,0.f,0.f,0.f};
    const __bf16* wp = wo_t + (size_t)(wave * 16 + lm) * 1024 + kb;
    const __bf16* xp = &Xls[lm << 10];
    #pragma unroll 4
    for (int kt = 0; kt < 32; ++kt) {
        const int k0 = kt * 32;
        v8bf lo = *(const v8bf*)(xp + k0 + ka);
        v8bf hg = *(const v8bf*)(xp + k0 + ka + 8);
        v16bf a = __builtin_shufflevector(lo, hg,
            0,1,2,3,4,5,6,7,8,9,10,11,12,13,14,15);
        v16bf b = *(const v16bf*)(wp + k0);
        po = __builtin_amdgcn_wmma_f32_16x16x32_bf16(
            false, a, false, b, (short)0, po, false, false);
    }

    // ---- epilogue: + bo, / (norm + 1e-3), write f32 [i, j, o] ----
    const int o = wave * 16 + lm;
    const float bov = bo[o];
    #pragma unroll
    for (int v = 0; v < 8; ++v) {
        const int pair = v + hi * 8;
        const int i = ibase + (pair >> 2);
        const int j = jbase + (pair & 3);
        const float nv = nrm[i * NN + j];
        out[((size_t)i * NN + j) * NOUT + o] = (po[v] + bov) / (nv + 0.001f);
    }
}

// ---------------------------------------------------------------------------
extern "C" void kernel_launch(void* const* d_in, const int* in_sizes, int n_in,
                              void* d_out, int out_size, void* d_ws, size_t ws_size,
                              hipStream_t stream) {
    const float* act  = (const float*)d_in[0];
    const float* mask = (const float*)d_in[1];
    const float* ln_g = (const float*)d_in[2];
    const float* ln_b = (const float*)d_in[3];
    const float* wl   = (const float*)d_in[4];
    const float* bl   = (const float*)d_in[5];
    const float* wr   = (const float*)d_in[6];
    const float* br   = (const float*)d_in[7];
    const float* wo   = (const float*)d_in[8];
    const float* bo   = (const float*)d_in[9];
    float* out = (float*)d_out;

    char* w = (char*)d_ws;
    const size_t LR_BYTES = (size_t)NP * NM * sizeof(__bf16);   // 8 MB each
    __bf16* Lbuf = (__bf16*)(w);
    __bf16* Rbuf = (__bf16*)(w + LR_BYTES);
    __bf16* wo_t = (__bf16*)(w + 2 * LR_BYTES);                 // 256 KB
    float*  nrmp = (float*)(w + 2 * LR_BYTES + (size_t)NOUT * 1024 * sizeof(__bf16));

    prep_wo_kernel<<<dim3(512), dim3(256), 0, stream>>>(wo, wo_t);
    norm_kernel<<<dim3(NN), dim3(NN), 0, stream>>>(mask, nrmp);
    ln_proj_kernel<<<dim3((NM * NN) / 32), dim3(256), 0, stream>>>(
        act, mask, ln_g, ln_b, wl, bl, wr, br, Lbuf, Rbuf);
    opm_gemm_kernel<<<dim3(NN / 4, NN / 4), dim3(256), 0, stream>>>(
        Lbuf, Rbuf, wo_t, bo, nrmp, out);
}
